// TopKRouter_61890478735807
// MI455X (gfx1250) — matrix operand order repaired and checked by reading, not compiled, not run on hardware
//
#include <hip/hip_runtime.h>
#include <hip/hip_bf16.h>

typedef __attribute__((ext_vector_type(16))) __bf16 v16bf;
typedef __attribute__((ext_vector_type(8)))  __bf16 v8bf;
typedef __attribute__((ext_vector_type(4)))  __bf16 v4bf;
typedef __attribute__((ext_vector_type(8)))  float  v8f;

#define H     4096
#define E     64
#define MTILE 128          // tokens per workgroup
#define KSTEP 32           // bf16 WMMA K
#define BSTR  40           // bf16 LDS row stride: 80B -> 16B aligned, conflict-free
#define LPAD  65           // logits row stride (odd dwords -> conflict-free)

// Split 4 fp32 -> 4 bf16 hi + 4 bf16 residual-lo (RNE), 8B each.
__device__ __forceinline__ void split4(float4 v, v4bf& hv, v4bf& lv) {
    float t[4] = {v.x, v.y, v.z, v.w};
#pragma unroll
    for (int j = 0; j < 4; ++j) {
        __bf16 h = (__bf16)t[j];
        hv[j] = h;
        lv[j] = (__bf16)(t[j] - (float)h);
    }
}

// Assemble a v16bf WMMA operand from two 16B LDS runs.
__device__ __forceinline__ v16bf pair16(const __bf16* p0, const __bf16* p1) {
    union { v16bf v; v8bf h[2]; } u;
    u.h[0] = *(const v8bf*)p0;
    u.h[1] = *(const v8bf*)p1;
    return u.v;
}

__global__ __launch_bounds__(256) void
topk_router_kernel(const float* __restrict__ X, const float* __restrict__ W,
                   float* __restrict__ w_out, float* __restrict__ i_out,
                   float* __restrict__ l_out) {
    __shared__ __align__(16) __bf16 xh[MTILE * BSTR];  // 10 KB  X hi plane
    __shared__ __align__(16) __bf16 xl[MTILE * BSTR];  // 10 KB  X lo plane
    __shared__ __align__(16) __bf16 wh[E * BSTR];      //  5 KB  W hi plane
    __shared__ __align__(16) __bf16 wl[E * BSTR];      //  5 KB  W lo plane
    __shared__ float ls[MTILE * LPAD];                 // 33 KB  logits staging

    const int tid  = threadIdx.x;
    const int lane = tid & 31;
    const int wv   = tid >> 5;           // 8 waves
    const int sel  = lane >> 4;          // 0: lanes 0-15, 1: lanes 16-31
    const int l16  = lane & 15;
    const int m0   = blockIdx.x * MTILE;

    // ISA 16-bit operand layouts (wave32):
    //   A 16x32: lanes 0-15 -> K 0..7 & 16..23 ; lanes 16-31 -> K 8..15 & 24..31
    //   B 32x16: lanes 0-15 -> K 0..7 & 16..23 ; lanes 16-31 -> K 8..15 & 24..31
    const int arow  = wv * 16 + l16;
    const int akoff = sel ? 8 : 0;
    const int bkoff = sel ? 8 : 0;

    // staging coordinates (fixed per thread)
    const int xrow[1] = {0};
    (void)xrow;

    v8f acc[4] = {};

    // ---- register pipeline: preload chunk k0 = 0 ----
    float4 xr[4], wr[2];
#pragma unroll
    for (int i = 0; i < 4; ++i) {
        int flat = i * 256 + tid, row = flat >> 3, c4 = (flat & 7) * 4;
        xr[i] = *(const float4*)(X + (size_t)(m0 + row) * H + c4);
    }
#pragma unroll
    for (int i = 0; i < 2; ++i) {
        int flat = i * 256 + tid, row = flat >> 3, c4 = (flat & 7) * 4;
        wr[i] = *(const float4*)(W + (size_t)row * H + c4);
    }

    for (int k0 = 0; k0 < H; k0 += KSTEP) {
        // ---- convert staged registers -> bf16 hi/lo planes in LDS ----
#pragma unroll
        for (int i = 0; i < 4; ++i) {
            int flat = i * 256 + tid, row = flat >> 3, c4 = (flat & 7) * 4;
            v4bf hv, lv;
            split4(xr[i], hv, lv);
            *(v4bf*)&xh[row * BSTR + c4] = hv;
            *(v4bf*)&xl[row * BSTR + c4] = lv;
        }
#pragma unroll
        for (int i = 0; i < 2; ++i) {
            int flat = i * 256 + tid, row = flat >> 3, c4 = (flat & 7) * 4;
            v4bf hv, lv;
            split4(wr[i], hv, lv);
            *(v4bf*)&wh[row * BSTR + c4] = hv;
            *(v4bf*)&wl[row * BSTR + c4] = lv;
        }
        __syncthreads();

        // ---- issue next chunk's global loads (overlap with WMMAs below) ----
        if (k0 + KSTEP < H) {
            int kn = k0 + KSTEP;
#pragma unroll
            for (int i = 0; i < 4; ++i) {
                int flat = i * 256 + tid, row = flat >> 3, c4 = (flat & 7) * 4;
                xr[i] = *(const float4*)(X + (size_t)(m0 + row) * H + kn + c4);
            }
#pragma unroll
            for (int i = 0; i < 2; ++i) {
                int flat = i * 256 + tid, row = flat >> 3, c4 = (flat & 7) * 4;
                wr[i] = *(const float4*)(W + (size_t)row * H + kn + c4);
            }
            if (kn + KSTEP < H)   // nudge the chunk after that toward L2
                __builtin_prefetch(X + (size_t)(m0 + (tid >> 3)) * H + kn + KSTEP, 0, 0);
        }

        // ---- pure ds_load -> WMMA compute, no conversions ----
        v16bf ah = pair16(&xh[arow * BSTR + akoff], &xh[arow * BSTR + akoff + 16]);
        v16bf al = pair16(&xl[arow * BSTR + akoff], &xl[arow * BSTR + akoff + 16]);

        v16bf bh[4], bl[4];
#pragma unroll
        for (int et = 0; et < 4; ++et) {
            int er = (et * 16 + l16) * BSTR + bkoff;
            bh[et] = pair16(&wh[er], &wh[er + 16]);
            bl[et] = pair16(&wl[er], &wl[er + 16]);
        }
        // term-major: 4 independent accumulators in flight per term
#pragma unroll
        for (int et = 0; et < 4; ++et)
            acc[et] = __builtin_amdgcn_wmma_f32_16x16x32_bf16(false, ah, false, bh[et],
                                                              (short)0, acc[et], false, false);
#pragma unroll
        for (int et = 0; et < 4; ++et)
            acc[et] = __builtin_amdgcn_wmma_f32_16x16x32_bf16(false, al, false, bh[et],
                                                              (short)0, acc[et], false, false);
#pragma unroll
        for (int et = 0; et < 4; ++et)
            acc[et] = __builtin_amdgcn_wmma_f32_16x16x32_bf16(false, ah, false, bl[et],
                                                              (short)0, acc[et], false, false);
        __syncthreads();
    }

    // ---- scatter accumulators to LDS (D layout: vgpr r -> rows r / r+8) ----
#pragma unroll
    for (int et = 0; et < 4; ++et)
#pragma unroll
        for (int r = 0; r < 8; ++r) {
            int trow = wv * 16 + sel * 8 + r;
            ls[trow * LPAD + et * 16 + l16] = acc[et][r];
        }
    __syncthreads();

    // ---- coalesced logits writeback ----
    for (int idx = tid; idx < MTILE * E; idx += 256) {
        int tl = idx >> 6, e = idx & 63;
        l_out[(size_t)(m0 + tl) * E + e] = ls[tl * LPAD + e];
    }

    // ---- top-2 + softmax over the pair ----
    if (tid < MTILE) {
        const float* row = &ls[tid * LPAD];
        float best = row[0], second = -3.402823466e38f;
        int bi = 0, si = 0;
#pragma unroll 4
        for (int e = 1; e < E; ++e) {
            float v = row[e];
            if (v > best)        { second = best; si = bi; best = v; bi = e; }
            else if (v > second) { second = v;    si = e; }
        }
        float p  = __expf(second - best);       // <= 1, no overflow
        float w0 = 1.0f / (1.0f + p);
        size_t t = (size_t)(m0 + tid);
        w_out[t * 2 + 0] = w0;
        w_out[t * 2 + 1] = 1.0f - w0;
        i_out[t * 2 + 0] = (float)bi;
        i_out[t * 2 + 1] = (float)si;
    }
}

extern "C" void kernel_launch(void* const* d_in, const int* in_sizes, int n_in,
                              void* d_out, int out_size, void* d_ws, size_t ws_size,
                              hipStream_t stream) {
    const float* X = (const float*)d_in[0];   // [4, 8192, 4096] fp32
    const float* W = (const float*)d_in[1];   // [64, 4096] fp32
    const int T = in_sizes[0] / H;            // 32768 tokens

    float* out   = (float*)d_out;
    float* w_out = out;                       // [T, 2]
    float* i_out = out + (size_t)T * 2;       // [T, 2] (indices as float)
    float* l_out = out + (size_t)T * 4;       // [T, 64]

    dim3 grid(T / MTILE), block(256);
    topk_router_kernel<<<grid, block, 0, stream>>>(X, W, w_out, i_out, l_out);
}
